// MHSA_48533130445460
// MI455X (gfx1250) — matrix operand order repaired and checked by reading, not compiled, not run on hardware
//
#include <hip/hip_runtime.h>

// ---------- types ----------
typedef __bf16 bf16_t;
typedef bf16_t       v16bf __attribute__((ext_vector_type(16)));
typedef float        v8f   __attribute__((ext_vector_type(8)));
typedef unsigned int u32x4 __attribute__((ext_vector_type(4)));

struct Pack32 { u32x4 lo, hi; };   // 32 bytes, trivially copyable

#define BATCH  4
#define SEQ    2048
#define DMODEL 1024
#define NHEAD  16
#define HDIM   64
#define NTOK   (BATCH * SEQ)       /* 8192 */
#define QKVN   (3 * DMODEL)        /* 3072 */

// ---------- WMMA helpers ----------
__device__ __forceinline__ v8f wmma_bf16(v16bf a, v16bf b, v8f c) {
  // D = A(16x32 bf16) * B(32x16 bf16) + C(16x16 f32)
  return __builtin_amdgcn_wmma_f32_16x16x32_bf16(false, a, false, b, (short)0, c,
                                                 false, false);
}

__device__ __forceinline__ v8f vzero8() {
  v8f z = {0.f, 0.f, 0.f, 0.f, 0.f, 0.f, 0.f, 0.f};
  return z;
}

__device__ __forceinline__ v16bf pack_frag(u32x4 lo, u32x4 hi) {
  Pack32 t; t.lo = lo; t.hi = hi;
  return __builtin_bit_cast(v16bf, t);
}

// A fragment (16x32, M x K), row-major source with leading dim `ld`.
// Lane layout (ISA 7.12.2): m = lane&15; lanes<16 hold k in [0,8)+[16,24),
// lanes>=16 hold k in [8,16)+[24,32). Two 16-byte contiguous chunks per lane.
__device__ __forceinline__ v16bf load_a(const bf16_t* __restrict__ base, int ld,
                                        int m0, int k0, int lane) {
  const int m = lane & 15, hi = lane >> 4;
  const bf16_t* p = base + (size_t)(m0 + m) * (size_t)ld + (size_t)(k0 + 8 * hi);
  u32x4 lo = *(const u32x4*)(p);        // k: base .. base+7
  u32x4 hh = *(const u32x4*)(p + 16);   // k: base+16 .. base+23
  return pack_frag(lo, hh);
}

// B fragment (32x16, K x N) loaded from a TRANSPOSED source laid out [N][K]
// row-major with leading dim `ld`. Lane layout: n = lane&15; lanes<16 hold k in [0,8)+[16,24),
// lanes>=16 hold k in [8,16)+[24,32). Two 16-byte contiguous chunks per lane.
__device__ __forceinline__ v16bf load_bT(const bf16_t* __restrict__ baseT, int ld,
                                         int n0, int k0, int lane) {
  const int n = lane & 15, hi = lane >> 4;
  const bf16_t* p = baseT + (size_t)(n0 + n) * (size_t)ld + (size_t)(k0 + 8 * hi);
  u32x4 lo = *(const u32x4*)(p);
  u32x4 hh = *(const u32x4*)(p + 16);
  return pack_frag(lo, hh);
}

// ---------- prep kernels ----------
__global__ __launch_bounds__(256) void f32_to_bf16_kernel(const float* __restrict__ in,
                                                          bf16_t* __restrict__ out,
                                                          int n) {
  int i = blockIdx.x * blockDim.x + threadIdx.x;
  if (i < n) out[i] = (bf16_t)in[i];
}

// out[n*rows + k] = (bf16) in[k*cols + n]   (transpose + convert)
__global__ __launch_bounds__(256) void transpose_to_bf16_kernel(const float* __restrict__ in,
                                                                bf16_t* __restrict__ out,
                                                                int rows, int cols) {
  __shared__ float tile[32][33];
  const int tx = threadIdx.x & 31;
  const int ty = threadIdx.x >> 5;           // 0..7
  const int k0 = blockIdx.x * 32;
  const int n0 = blockIdx.y * 32;
  #pragma unroll
  for (int r = ty; r < 32; r += 8)
    tile[r][tx] = in[(size_t)(k0 + r) * cols + (n0 + tx)];
  __syncthreads();
  #pragma unroll
  for (int r = ty; r < 32; r += 8)
    out[(size_t)(n0 + r) * rows + (k0 + tx)] = (bf16_t)tile[tx][r];
}

// ---------- QKV GEMM:  qkv = x @ W_qkv + b  -> Q, K (row-major), V (transposed) ----------
__global__ __launch_bounds__(128) void qkv_gemm_kernel(
    const bf16_t* __restrict__ X,    // [NTOK][DMODEL] bf16
    const bf16_t* __restrict__ WT,   // [QKVN][DMODEL] bf16 (W_qkv transposed)
    const float* __restrict__ bias,  // [QKVN] f32
    bf16_t* __restrict__ Q,          // [B,H,SEQ,HDIM]
    bf16_t* __restrict__ K,          // [B,H,SEQ,HDIM]
    bf16_t* __restrict__ Vt)         // [B,H,HDIM,SEQ]
{
  const int lane = threadIdx.x & 31;
  const int wave = threadIdx.x >> 5;
  const int m0 = blockIdx.x * 128 + wave * 32;   // token tile
  const int n0 = blockIdx.y * 64;                // output-column tile (one head slab)

  v8f acc[2][4];
  #pragma unroll
  for (int i = 0; i < 2; ++i)
    #pragma unroll
    for (int j = 0; j < 4; ++j) acc[i][j] = vzero8();

  for (int k0 = 0; k0 < DMODEL; k0 += 32) {
    v16bf a0 = load_a(X, DMODEL, m0,      k0, lane);
    v16bf a1 = load_a(X, DMODEL, m0 + 16, k0, lane);
    #pragma unroll
    for (int j = 0; j < 4; ++j) {
      v16bf b = load_bT(WT, DMODEL, n0 + j * 16, k0, lane);
      acc[0][j] = wmma_bf16(a0, b, acc[0][j]);
      acc[1][j] = wmma_bf16(a1, b, acc[1][j]);
    }
  }

  const int nlo = lane & 15, hi = lane >> 4;
  const int c = n0 / DMODEL;              // 0=q 1=k 2=v (constant per wave)
  const int h = (n0 % DMODEL) / HDIM;     // head (constant per wave; n0 % 64 == 0)
  #pragma unroll
  for (int i = 0; i < 2; ++i) {
    #pragma unroll
    for (int j = 0; j < 4; ++j) {
      const int   n  = n0 + j * 16 + nlo;
      const float bv = bias[n];
      const int   d  = j * 16 + nlo;      // 0..63 within head
      #pragma unroll
      for (int v = 0; v < 8; ++v) {
        const int m  = m0 + i * 16 + v + 8 * hi;
        const int bb = m >> 11;           // / SEQ
        const int t  = m & (SEQ - 1);
        const bf16_t ov = (bf16_t)(acc[i][j][v] + bv);
        const size_t head = (size_t)(bb * NHEAD + h);
        if (c == 0)      Q [(head * SEQ  + t) * HDIM + d] = ov;
        else if (c == 1) K [(head * SEQ  + t) * HDIM + d] = ov;
        else             Vt[(head * HDIM + d) * SEQ  + t] = ov;
      }
    }
  }
}

// ---------- Flash attention: per wave one (b,h, 16-row q tile) ----------
__global__ __launch_bounds__(128) void attn_kernel(
    const bf16_t* __restrict__ Q,   // [B*H][SEQ][HDIM]
    const bf16_t* __restrict__ K,   // [B*H][SEQ][HDIM]
    const bf16_t* __restrict__ Vt,  // [B*H][HDIM][SEQ]
    bf16_t* __restrict__ Y)         // [B][SEQ][DMODEL]
{
  __shared__ bf16_t ptile[4][16][32];   // per-wave 16x32 P tile (C-layout -> A-layout)
  const int lane = threadIdx.x & 31;
  const int wave = threadIdx.x >> 5;
  const int nlo = lane & 15, hi = lane >> 4;
  const int task = blockIdx.x * 4 + wave;   // 0..8191
  const int qt = task & 127;
  const int bh = task >> 7;                 // 0..63
  const bf16_t* Qh = Q  + (size_t)bh * SEQ  * HDIM;
  const bf16_t* Kh = K  + (size_t)bh * SEQ  * HDIM;
  const bf16_t* Vh = Vt + (size_t)bh * HDIM * SEQ;
  const int q0 = qt * 16;

  const v16bf aq0 = load_a(Qh, HDIM, q0, 0,  lane);
  const v16bf aq1 = load_a(Qh, HDIM, q0, 32, lane);

  v8f o[4];
  #pragma unroll
  for (int j = 0; j < 4; ++j) o[j] = vzero8();
  float mrow[8], lrow[8];
  #pragma unroll
  for (int v = 0; v < 8; ++v) { mrow[v] = -3.0e38f; lrow[v] = 0.f; }
  const float scale = 0.125f;   // 1/sqrt(64)

  for (int kt0 = 0; kt0 < SEQ; kt0 += 32) {
    // S = scale * Q Kt  for 32 keys (two 16-wide C tiles)
    v8f s0 = vzero8(), s1 = vzero8();
    {
      v16bf b00 = load_bT(Kh, HDIM, kt0,      0,  lane);
      v16bf b01 = load_bT(Kh, HDIM, kt0,      32, lane);
      v16bf b10 = load_bT(Kh, HDIM, kt0 + 16, 0,  lane);
      v16bf b11 = load_bT(Kh, HDIM, kt0 + 16, 32, lane);
      s0 = wmma_bf16(aq0, b00, s0);
      s0 = wmma_bf16(aq1, b01, s0);
      s1 = wmma_bf16(aq0, b10, s1);
      s1 = wmma_bf16(aq1, b11, s1);
    }
    // online softmax: rows live across lanes 0-15 / 16-31 halves (xor<16 stays in half)
    float tmax[8];
    #pragma unroll
    for (int v = 0; v < 8; ++v) tmax[v] = fmaxf(s0[v], s1[v]) * scale;
    #pragma unroll
    for (int off = 1; off < 16; off <<= 1) {
      #pragma unroll
      for (int v = 0; v < 8; ++v)
        tmax[v] = fmaxf(tmax[v], __shfl_xor(tmax[v], off, 32));
    }
    float corr[8], rsum[8];
    #pragma unroll
    for (int v = 0; v < 8; ++v) {
      const float mn = fmaxf(mrow[v], tmax[v]);
      corr[v] = __expf(mrow[v] - mn);
      mrow[v] = mn;
      const float p0 = __expf(s0[v] * scale - mn);
      const float p1 = __expf(s1[v] * scale - mn);
      s0[v] = p0; s1[v] = p1;
      rsum[v] = p0 + p1;
    }
    #pragma unroll
    for (int off = 1; off < 16; off <<= 1) {
      #pragma unroll
      for (int v = 0; v < 8; ++v) rsum[v] += __shfl_xor(rsum[v], off, 32);
    }
    #pragma unroll
    for (int v = 0; v < 8; ++v) lrow[v] = lrow[v] * corr[v] + rsum[v];
    #pragma unroll
    for (int j = 0; j < 4; ++j)
      #pragma unroll
      for (int v = 0; v < 8; ++v) o[j][v] = o[j][v] * corr[v];

    // C-layout -> A-layout through per-wave LDS tile (DS ops are in-order per wave)
    #pragma unroll
    for (int v = 0; v < 8; ++v) {
      ptile[wave][v + 8 * hi][nlo]      = (bf16_t)s0[v];
      ptile[wave][v + 8 * hi][nlo + 16] = (bf16_t)s1[v];
    }
    asm volatile("s_wait_dscnt 0" ::: "memory");
    const bf16_t* pr = &ptile[wave][nlo][8 * hi];
    u32x4 plo = *(const u32x4*)(pr);
    u32x4 phi = *(const u32x4*)(pr + 16);
    asm volatile("" ::: "memory");
    v16bf pf = pack_frag(plo, phi);

    // O += P (16x32) * V (32x64)
    #pragma unroll
    for (int j = 0; j < 4; ++j) {
      v16bf bv = load_bT(Vh, SEQ, j * 16, kt0, lane);
      o[j] = wmma_bf16(pf, bv, o[j]);
    }
  }

  // finalize: O / l  -> Y[b, t, h*64 + d]  (bf16, feeds proj GEMM as A-matrix)
  const int bb = bh >> 4, h = bh & 15;
  #pragma unroll
  for (int j = 0; j < 4; ++j) {
    #pragma unroll
    for (int v = 0; v < 8; ++v) {
      const int t = q0 + v + 8 * hi;
      const float val = o[j][v] / lrow[v];
      Y[((size_t)(bb * SEQ + t)) * DMODEL + h * HDIM + j * 16 + nlo] = (bf16_t)val;
    }
  }
}

// ---------- Output projection: out = Y @ W_proj + b_proj (f32 out) ----------
__global__ __launch_bounds__(128) void proj_gemm_kernel(
    const bf16_t* __restrict__ Yb,   // [NTOK][DMODEL] bf16
    const bf16_t* __restrict__ WT,   // [DMODEL][DMODEL] bf16 (W_proj transposed)
    const float* __restrict__ bias,  // [DMODEL]
    float* __restrict__ out)         // [NTOK][DMODEL] f32
{
  const int lane = threadIdx.x & 31;
  const int wave = threadIdx.x >> 5;
  const int m0 = blockIdx.x * 128 + wave * 32;
  const int n0 = blockIdx.y * 64;

  v8f acc[2][4];
  #pragma unroll
  for (int i = 0; i < 2; ++i)
    #pragma unroll
    for (int j = 0; j < 4; ++j) acc[i][j] = vzero8();

  for (int k0 = 0; k0 < DMODEL; k0 += 32) {
    v16bf a0 = load_a(Yb, DMODEL, m0,      k0, lane);
    v16bf a1 = load_a(Yb, DMODEL, m0 + 16, k0, lane);
    #pragma unroll
    for (int j = 0; j < 4; ++j) {
      v16bf b = load_bT(WT, DMODEL, n0 + j * 16, k0, lane);
      acc[0][j] = wmma_bf16(a0, b, acc[0][j]);
      acc[1][j] = wmma_bf16(a1, b, acc[1][j]);
    }
  }

  const int nlo = lane & 15, hi = lane >> 4;
  #pragma unroll
  for (int i = 0; i < 2; ++i) {
    #pragma unroll
    for (int j = 0; j < 4; ++j) {
      const int n = n0 + j * 16 + nlo;
      const float bv = bias[n];
      #pragma unroll
      for (int v = 0; v < 8; ++v) {
        const int m = m0 + i * 16 + v + 8 * hi;
        out[(size_t)m * DMODEL + n] = acc[i][j][v] + bv;
      }
    }
  }
}

// ---------- host launcher ----------
extern "C" void kernel_launch(void* const* d_in, const int* in_sizes, int n_in,
                              void* d_out, int out_size, void* d_ws, size_t ws_size,
                              hipStream_t stream) {
  (void)in_sizes; (void)n_in; (void)out_size; (void)ws_size;
  const float* x     = (const float*)d_in[0];   // [4,2048,1024]
  const float* Wqkv  = (const float*)d_in[1];   // [1024,3072]
  const float* bqkv  = (const float*)d_in[2];   // [3072]
  const float* Wproj = (const float*)d_in[3];   // [1024,1024]
  const float* bproj = (const float*)d_in[4];   // [1024]
  float* out = (float*)d_out;

  char* ws = (char*)d_ws;
  size_t off = 0;
  bf16_t* xb     = (bf16_t*)(ws + off); off += (size_t)NTOK * DMODEL * 2;       // 16 MB
  bf16_t* WqkvT  = (bf16_t*)(ws + off); off += (size_t)QKVN * DMODEL * 2;       //  6 MB
  bf16_t* WprojT = (bf16_t*)(ws + off); off += (size_t)DMODEL * DMODEL * 2;     //  2 MB
  bf16_t* Qb     = (bf16_t*)(ws + off); off += (size_t)NTOK * DMODEL * 2;       // 16 MB
  bf16_t* Kb     = (bf16_t*)(ws + off); off += (size_t)NTOK * DMODEL * 2;       // 16 MB
  bf16_t* Vt     = (bf16_t*)(ws + off); off += (size_t)NTOK * DMODEL * 2;       // 16 MB
  bf16_t* Yb     = (bf16_t*)(ws + off); off += (size_t)NTOK * DMODEL * 2;       // 16 MB

  const int nx = NTOK * DMODEL;
  f32_to_bf16_kernel<<<(nx + 255) / 256, 256, 0, stream>>>(x, xb, nx);
  transpose_to_bf16_kernel<<<dim3(DMODEL / 32, QKVN / 32),  256, 0, stream>>>(Wqkv,  WqkvT,  DMODEL, QKVN);
  transpose_to_bf16_kernel<<<dim3(DMODEL / 32, DMODEL / 32), 256, 0, stream>>>(Wproj, WprojT, DMODEL, DMODEL);

  qkv_gemm_kernel<<<dim3(NTOK / 128, QKVN / 64), 128, 0, stream>>>(xb, WqkvT, bqkv, Qb, Kb, Vt);
  attn_kernel<<<(BATCH * NHEAD * (SEQ / 16)) / 4, 128, 0, stream>>>(Qb, Kb, Vt, Yb);
  proj_gemm_kernel<<<dim3(NTOK / 128, DMODEL / 64), 128, 0, stream>>>(Yb, WprojT, bproj, out);
}
